// SelfAttentionBaseline_44452911514380
// MI455X (gfx1250) — matrix operand order repaired and checked by reading, not compiled, not run on hardware
//
#include <hip/hip_runtime.h>

typedef __attribute__((ext_vector_type(16))) __bf16 v16bf;
typedef __attribute__((ext_vector_type(8)))  float  v8f;
typedef __attribute__((ext_vector_type(4)))  int    v4i;
typedef unsigned short u16;
typedef unsigned int   u32;

#if __has_builtin(__builtin_amdgcn_global_load_async_to_lds_b128)
#define USE_ASYNC_LDS 1
#define ASYNC_LD_B128(g, l)                                          \
  __builtin_amdgcn_global_load_async_to_lds_b128(                    \
      (__attribute__((address_space(1))) v4i*)((v4i*)(g)),           \
      (__attribute__((address_space(3))) v4i*)((v4i*)(l)), 0, 0)
#else
#define USE_ASYNC_LDS 0
#endif

#if __has_builtin(__builtin_amdgcn_s_wait_asynccnt)
#define WAIT_ASYNC() __builtin_amdgcn_s_wait_asynccnt(0)
#else
#define WAIT_ASYNC() asm volatile("s_wait_asynccnt 0x0" ::: "memory")
#endif

__device__ __forceinline__ u16 f2bf(float f) {
  u32 u = __float_as_uint(f);
  u32 r = u + 0x7FFFu + ((u >> 16) & 1u);   // round-to-nearest-even
  return (u16)(r >> 16);
}

union Frag  { uint4 q[2]; u16 us[16]; v16bf bf; };
union Pack8 { uint4 q;    u16 us[8]; };

// ---------------------------------------------------------------------------
// Elementwise f32 -> bf16 convert (vectorized x4)
// ---------------------------------------------------------------------------
__global__ __launch_bounds__(256) void cvt_bf16(const float4* __restrict__ in,
                                                uint2* __restrict__ out, long n4) {
  long i = (long)blockIdx.x * 256 + threadIdx.x;
  if (i < n4) {
    float4 v = in[i];
    uint2 r;
    r.x = (u32)f2bf(v.x) | ((u32)f2bf(v.y) << 16);
    r.y = (u32)f2bf(v.z) | ((u32)f2bf(v.w) << 16);
    out[i] = r;
  }
}

// ---------------------------------------------------------------------------
// Transpose + convert: out[c][r] = bf16(in[r][c]); in is R x C f32.
// ---------------------------------------------------------------------------
__global__ __launch_bounds__(256) void transpose_cvt(const float* __restrict__ in,
                                                     u16* __restrict__ out,
                                                     int R, int C) {
  __shared__ float tile[32][33];
  int tx = threadIdx.x, ty = threadIdx.y;
  int c0 = blockIdx.x * 32, r0 = blockIdx.y * 32;
#pragma unroll
  for (int j = 0; j < 4; ++j)
    tile[ty + 8 * j][tx] = in[(long)(r0 + ty + 8 * j) * C + c0 + tx];
  __syncthreads();
#pragma unroll
  for (int j = 0; j < 4; ++j)
    out[(long)(c0 + ty + 8 * j) * R + r0 + tx] = f2bf(tile[tx][ty + 8 * j]);
}

// ---------------------------------------------------------------------------
// Tiled WMMA GEMM:  C[M][N] = scale * A[M][K] x Bt[N][K]^T  (bf16 in, f32 acc)
//   OUT_MODE 0: f32 row-major    OUT_MODE 1: bf16 row-major
//   OUT_MODE 2: bf16 TRANSPOSED  (stores Ct[N][M], ldc = M-stride)
// Block = 256 threads (8 waves). Block tile 128x128, K stepped by 32.
// Wave w: rows [32*(w&3), +32), cols [64*(w>>2), +64) -> 2x4 = 8 WMMA C-tiles.
// Double-buffered LDS, software-pipelined global->LDS (async if available).
// ---------------------------------------------------------------------------
template <int OUT_MODE>
__global__ __launch_bounds__(256) void gemm_bf16(
    const u16* __restrict__ A, const u16* __restrict__ Bt, void* __restrict__ C,
    int K, int lda, int ldb, int ldc,
    long sA, long sB, long sC, float scale) {
  int b = blockIdx.z;
  A  += (long)b * sA;
  Bt += (long)b * sB;
  float* Cf = (float*)C + (long)b * sC;
  u16*   Cb = (u16*)C   + (long)b * sC;

  __shared__ __align__(16) u16 As[2][128 * 32];
  __shared__ __align__(16) u16 Bs[2][128 * 32];

  const int m0 = blockIdx.y * 128, n0 = blockIdx.x * 128;
  const int t = threadIdx.x;
  const int w = t >> 5, lane = t & 31;
  const int hf = lane >> 4, col = lane & 15;
  const int wm = w & 3, wn = w >> 2;

  const int lrow = t >> 2;       // 0..63 (+64 for second half)
  const int lcol = (t & 3) * 8;  // 0,8,16,24

  const u16* Ag0 = A  + (long)(m0 + lrow) * lda + lcol;
  const u16* Ag1 = A  + (long)(m0 + lrow + 64) * lda + lcol;
  const u16* Bg0 = Bt + (long)(n0 + lrow) * ldb + lcol;
  const u16* Bg1 = Bt + (long)(n0 + lrow + 64) * ldb + lcol;

  v8f acc[2][4];
#pragma unroll
  for (int mi = 0; mi < 2; ++mi)
#pragma unroll
    for (int ni = 0; ni < 4; ++ni) acc[mi][ni] = (v8f){};

#if !USE_ASYNC_LDS
  uint4 ra0, ra1, rb0, rb1;
#endif

  auto compute = [&](int buf) {
    const u16* Ab = &As[buf][0];
    const u16* Bb = &Bs[buf][0];
    Frag fa[2], fb[4];
#pragma unroll
    for (int mi = 0; mi < 2; ++mi) {
      const u16* ar = Ab + (32 * wm + 16 * mi + col) * 32;
      fa[mi].q[0] = *(const uint4*)(ar + hf * 8);
      fa[mi].q[1] = *(const uint4*)(ar + 16 + hf * 8);
    }
#pragma unroll
    for (int ni = 0; ni < 4; ++ni) {
      const u16* br = Bb + (64 * wn + 16 * ni + col) * 32;
      fb[ni].q[0] = *(const uint4*)(br + hf * 8);
      fb[ni].q[1] = *(const uint4*)(br + 16 + hf * 8);
    }
#pragma unroll
    for (int mi = 0; mi < 2; ++mi)
#pragma unroll
      for (int ni = 0; ni < 4; ++ni)
        acc[mi][ni] = __builtin_amdgcn_wmma_f32_16x16x32_bf16(
            false, fa[mi].bf, false, fb[ni].bf, (short)0, acc[mi][ni], false,
            false);
  };

#if USE_ASYNC_LDS
  auto prefetch = [&](int k0, int buf) {
    ASYNC_LD_B128(Ag0 + k0, &As[buf][lrow * 32 + lcol]);
    ASYNC_LD_B128(Ag1 + k0, &As[buf][(lrow + 64) * 32 + lcol]);
    ASYNC_LD_B128(Bg0 + k0, &Bs[buf][lrow * 32 + lcol]);
    ASYNC_LD_B128(Bg1 + k0, &Bs[buf][(lrow + 64) * 32 + lcol]);
  };
  prefetch(0, 0);
  WAIT_ASYNC();
  __syncthreads();
  int cur = 0;
  for (int k0 = 0; k0 < K; k0 += 32) {
    int nxt = cur ^ 1;
    if (k0 + 32 < K) prefetch(k0 + 32, nxt);
    compute(cur);
    WAIT_ASYNC();
    __syncthreads();
    cur = nxt;
  }
#else
  auto load_regs = [&](int k0) {
    ra0 = *(const uint4*)(Ag0 + k0);
    ra1 = *(const uint4*)(Ag1 + k0);
    rb0 = *(const uint4*)(Bg0 + k0);
    rb1 = *(const uint4*)(Bg1 + k0);
  };
  auto store_lds = [&](int buf) {
    *(uint4*)(&As[buf][lrow * 32 + lcol])        = ra0;
    *(uint4*)(&As[buf][(lrow + 64) * 32 + lcol]) = ra1;
    *(uint4*)(&Bs[buf][lrow * 32 + lcol])        = rb0;
    *(uint4*)(&Bs[buf][(lrow + 64) * 32 + lcol]) = rb1;
  };
  load_regs(0);
  store_lds(0);
  __syncthreads();
  int cur = 0;
  for (int k0 = 0; k0 < K; k0 += 32) {
    int nxt = cur ^ 1;
    bool more = (k0 + 32 < K);
    if (more) load_regs(k0 + 32);   // issue early; wait deferred to store
    compute(cur);
    if (more) store_lds(nxt);
    __syncthreads();
    cur = nxt;
  }
#endif

  const int gm = m0 + 32 * wm;
  const int gn = n0 + 64 * wn;

  if (OUT_MODE == 0) {
#pragma unroll
    for (int mi = 0; mi < 2; ++mi)
#pragma unroll
      for (int ni = 0; ni < 4; ++ni)
#pragma unroll
        for (int r = 0; r < 8; ++r) {
          int mm = gm + 16 * mi + r + 8 * hf;
          Cf[(long)mm * ldc + gn + 16 * ni + col] = acc[mi][ni][r] * scale;
        }
  } else if (OUT_MODE == 1) {
#pragma unroll
    for (int mi = 0; mi < 2; ++mi)
#pragma unroll
      for (int ni = 0; ni < 4; ++ni)
#pragma unroll
        for (int r = 0; r < 8; ++r) {
          int mm = gm + 16 * mi + r + 8 * hf;
          Cb[(long)mm * ldc + gn + 16 * ni + col] = f2bf(acc[mi][ni][r] * scale);
        }
  } else {
    // transposed: lane owns column n, contiguous 8 rows at gm + 16*mi + hf*8
#pragma unroll
    for (int mi = 0; mi < 2; ++mi)
#pragma unroll
      for (int ni = 0; ni < 4; ++ni) {
        Pack8 p;
#pragma unroll
        for (int r = 0; r < 8; ++r) p.us[r] = f2bf(acc[mi][ni][r] * scale);
        *(uint4*)(Cb + (long)(gn + 16 * ni + col) * ldc + gm + 16 * mi +
                  hf * 8) = p.q;
      }
  }
}

// ---------------------------------------------------------------------------
// Row softmax: one block per row; f32 in, bf16 out.
// ---------------------------------------------------------------------------
__global__ __launch_bounds__(256) void softmax_rows(const float* __restrict__ Sc,
                                                    u16* __restrict__ P, int n) {
  long row = blockIdx.x;
  const float* x = Sc + row * (long)n;
  u16* p = P + row * (long)n;
  int t = threadIdx.x, lane = t & 31, w = t >> 5;
  __shared__ float red[8];

  float lm = -3.4e38f;
  for (int i = t; i < n; i += 256) lm = fmaxf(lm, x[i]);
#pragma unroll
  for (int o = 16; o; o >>= 1) lm = fmaxf(lm, __shfl_xor(lm, o, 32));
  if (!lane) red[w] = lm;
  __syncthreads();
  float m = fmaxf(fmaxf(fmaxf(red[0], red[1]), fmaxf(red[2], red[3])),
                  fmaxf(fmaxf(red[4], red[5]), fmaxf(red[6], red[7])));
  __syncthreads();

  float ls = 0.f;
  for (int i = t; i < n; i += 256) ls += __expf(x[i] - m);
#pragma unroll
  for (int o = 16; o; o >>= 1) ls += __shfl_xor(ls, o, 32);
  if (!lane) red[w] = ls;
  __syncthreads();
  float tot = red[0] + red[1] + red[2] + red[3] + red[4] + red[5] + red[6] + red[7];
  float inv = 1.f / tot;

  for (int i = t; i < n; i += 256) p[i] = f2bf(__expf(x[i] - m) * inv);
}

// ---------------------------------------------------------------------------
extern "C" void kernel_launch(void* const* d_in, const int* in_sizes, int n_in,
                              void* d_out, int out_size, void* d_ws, size_t ws_size,
                              hipStream_t stream) {
  const int B = 4, S = 2048, D = 1024;
  const int T = B * S;  // 8192 tokens
  const float* X  = (const float*)d_in[0];
  const float* Wq = (const float*)d_in[1];
  const float* Wk = (const float*)d_in[2];
  const float* Wv = (const float*)d_in[3];
  const float* Wo = (const float*)d_in[4];
  float* Y = (float*)d_out;

  char* ws = (char*)d_ws;
  size_t off = 0;
  auto alloc = [&](size_t bytes) -> void* {
    void* p = ws + off;
    off += (bytes + 255) & ~(size_t)255;
    return p;
  };
  u16* Xb  = (u16*)alloc((size_t)T * D * 2);
  u16* WqT = (u16*)alloc((size_t)D * D * 2);
  u16* WkT = (u16*)alloc((size_t)D * D * 2);
  u16* WvT = (u16*)alloc((size_t)D * D * 2);
  u16* WoT = (u16*)alloc((size_t)D * D * 2);
  u16* Qb  = (u16*)alloc((size_t)T * D * 2);
  u16* Kb  = (u16*)alloc((size_t)T * D * 2);
  u16* Vt  = (u16*)alloc((size_t)D * T * 2);           // Vt[dv][token]
  u16* Hb  = (u16*)alloc((size_t)T * D * 2);
  u16* Pp  = (u16*)alloc((size_t)B * S * S * 2);
  float* Sc = (float*)alloc((size_t)B * S * S * 4);

  // 1) X -> bf16
  cvt_bf16<<<(T * D / 4 + 255) / 256, 256, 0, stream>>>(
      (const float4*)X, (uint2*)Xb, (long)T * D / 4);

  // 2) weights -> bf16, transposed (Bt[n][k] layout)
  dim3 tb(32, 8), tg(D / 32, D / 32);
  transpose_cvt<<<tg, tb, 0, stream>>>(Wq, WqT, D, D);
  transpose_cvt<<<tg, tb, 0, stream>>>(Wk, WkT, D, D);
  transpose_cvt<<<tg, tb, 0, stream>>>(Wv, WvT, D, D);
  transpose_cvt<<<tg, tb, 0, stream>>>(Wo, WoT, D, D);

  // 3) projections: Q,K row-major bf16; V transposed bf16
  dim3 gproj(D / 128, T / 128, 1);
  gemm_bf16<1><<<gproj, 256, 0, stream>>>(Xb, WqT, Qb, D, D, D, D, 0, 0, 0, 1.f);
  gemm_bf16<1><<<gproj, 256, 0, stream>>>(Xb, WkT, Kb, D, D, D, D, 0, 0, 0, 1.f);
  gemm_bf16<2><<<gproj, 256, 0, stream>>>(Xb, WvT, Vt, D, D, D, /*ldc=*/T, 0, 0, 0, 1.f);

  // 4) scores = (1/sqrt(D)) * Q K^T   (per batch via grid.z; K row-major == Bt)
  dim3 gsc(S / 128, S / 128, B);
  gemm_bf16<0><<<gsc, 256, 0, stream>>>(Qb, Kb, Sc, D, D, D, S,
                                        (long)S * D, (long)S * D, (long)S * S,
                                        0.03125f);

  // 5) softmax rows -> P (bf16)
  softmax_rows<<<B * S, 256, 0, stream>>>(Sc, Pp, S);

  // 6) H = P x V   (Bt = Vt with row stride T, batch offset b*S inside rows)
  dim3 gpv(D / 128, S / 128, B);
  gemm_bf16<1><<<gpv, 256, 0, stream>>>(Pp, Vt, Hb, S, S, T, D,
                                        (long)S * S, (long)S, (long)S * D, 1.f);

  // 7) y = H x W_O  (f32 out)
  dim3 gout(D / 128, T / 128, 1);
  gemm_bf16<0><<<gout, 256, 0, stream>>>(Hb, WoT, Y, D, D, D, D, 0, 0, 0, 1.f);
}
